// MolGCN_36644660969535
// MI455X (gfx1250) — matrix operand order repaired and checked by reading, not compiled, not run on hardware
//
#include <hip/hip_runtime.h>

typedef __attribute__((ext_vector_type(16))) _Float16 v16h;
typedef __attribute__((ext_vector_type(8)))  _Float16 v8h;
typedef __attribute__((ext_vector_type(8)))  float    v8f;

#define LN_EPS 1e-5f

// ---------------------------------------------------------------------------
// WMMA helpers (gfx1250, wave32). Layouts used here (A/B take a contiguous k order, not the ISA table's):
//  A (16x32 f16): lane L: row M=L&15, half kh=L>>4;
//    elem j -> K = kb + kh*16 + j (contiguous, matches B)
//  B (32x16 f16): lane L: col N=L&15, koff=(L>>4)*16; elem e -> K = kb+koff+e
//  C/D (16x16 f32): lane L: col N=L&15; vgpr r -> row M = r + 8*(L>>4)
// B is staged into LDS pre-swizzled: one contiguous v16h (32B) per (frag,lane),
// so each fragment load is 2x ds_load_b128 instead of 16x ds_load_u16.
// ---------------------------------------------------------------------------
__device__ __forceinline__ v8f wmma_f16(v16h a, v16h b, v8f c) {
  return __builtin_amdgcn_wmma_f32_16x16x32_f16(
      /*neg_a=*/false, a, /*neg_b=*/false, b,
      /*c_mod=*/(short)0, c, /*reuse_a=*/false, /*reuse_b=*/false);
}

__device__ __forceinline__ v16h load_A_frag(const _Float16* sh, int ld, int kb, int lane) {
  int m  = lane & 15;
  int kh = (lane >> 4) & 1;
  const v8h a0 = *(const v8h*)(sh + m * ld + kb + kh * 16);
  const v8h a1 = *(const v8h*)(sh + m * ld + kb + kh * 16 + 8);
  return __builtin_shufflevector(a0, a1, 0, 1, 2, 3, 4, 5, 6, 7,
                                 8, 9, 10, 11, 12, 13, 14, 15);
}

// Stage a [KT*32][128] f32 weight into LDS as swizzled f16 B-fragments.
// Fragment (kt, nt) occupies sW[((kt<<3)|nt)*32 + lane]; lane's 16 elems are
// W[(kt*32 + (lane>>4)*16 + e)*128 + nt*16 + (lane&15)], e = 0..15.
template <int KT>
__device__ __forceinline__ void stage_B(const float* __restrict__ W, v16h* sW, int tid) {
  const int lane = tid & 31, wv = tid >> 5;
  const int n = lane & 15, khi = lane >> 4;
  for (int frag = wv; frag < KT * 8; frag += 8) {
    int kt = frag >> 3, nt = frag & 7;
    int kb = kt * 32 + khi * 16;
    int nb = nt * 16;
    v16h v;
#pragma unroll
    for (int e = 0; e < 16; ++e) v[e] = (_Float16)W[(size_t)(kb + e) * 128 + nb + n];
    sW[frag * 32 + lane] = v;
  }
}

__device__ __forceinline__ v16h load_B_sw(const v16h* sW, int kt, int nt, int lane) {
  return sW[(((kt << 3) | nt) << 5) + lane];
}

// ---------------------------------------------------------------------------
// zero-fill (hipMemsetAsync avoided for graph-capture safety)
// ---------------------------------------------------------------------------
__global__ void zero_kernel(float* __restrict__ p, long n) {
  long i = (long)blockIdx.x * blockDim.x + threadIdx.x;
  long stride = (long)gridDim.x * blockDim.x;
  for (; i < n; i += stride) p[i] = 0.0f;
}

// ---------------------------------------------------------------------------
// Edge message + scatter:  aggr[dst] += relu(x[src] + ea @ We + be)
// One wave (32 lanes) per edge, D/32 features per lane, f32 atomics (L2-resident).
// ---------------------------------------------------------------------------
template <int D>
__global__ __launch_bounds__(256) void gine_msg_kernel(
    const float* __restrict__ x, const int* __restrict__ src, const int* __restrict__ dst,
    const float* __restrict__ ea, const float* __restrict__ We, const float* __restrict__ be,
    float* __restrict__ aggr, int nEdges) {
  __shared__ float sWe[16 * D];
  __shared__ float sBe[D];
  const int tid = threadIdx.x, lane = tid & 31, wv = tid >> 5;
  for (int i = tid; i < 16 * D; i += 256) sWe[i] = We[i];
  for (int i = tid; i < D; i += 256) sBe[i] = be[i];
  __syncthreads();

  constexpr int FPL = D / 32;
  const int nw = gridDim.x * 8;
  for (int e = blockIdx.x * 8 + wv; e < nEdges; e += nw) {
    const int s = src[e], d = dst[e];
    // prefetch next edge's source row (global_prefetch_b8 on gfx1250)
    int e2 = e + nw;
    if (e2 < nEdges) __builtin_prefetch(&x[(size_t)src[e2] * D], 0, 0);

    float eav = (lane < 16) ? ea[(size_t)e * 16 + lane] : 0.0f;
    float acc[FPL];
#pragma unroll
    for (int j = 0; j < FPL; ++j) acc[j] = sBe[lane + 32 * j];
#pragma unroll
    for (int k = 0; k < 16; ++k) {
      float ak = __shfl(eav, k, 32);
#pragma unroll
      for (int j = 0; j < FPL; ++j) acc[j] += ak * sWe[k * D + lane + 32 * j];
    }
#pragma unroll
    for (int j = 0; j < FPL; ++j) {
      float m = fmaxf(x[(size_t)s * D + lane + 32 * j] + acc[j], 0.0f);
      atomicAdd(&aggr[(size_t)d * D + lane + 32 * j], m);
    }
  }
}

// ---------------------------------------------------------------------------
// Node phase: h = x + aggr;  t = relu(h@W1+b1);  u = relu(t@W2+b2);
//             out = LayerNorm(u + res) with res = (PROJ ? x@Wr+br : x)
// 1 workgroup = 16 nodes, 8 waves each own one 16-col WMMA output tile.
// Single 32KB swizzled LDS weight buffer reused for Wr -> W1 -> W2.
// ---------------------------------------------------------------------------
template <int DIN, bool PROJ>
__global__ __launch_bounds__(256) void gine_node_kernel(
    const float* __restrict__ xin, const float* __restrict__ aggr,
    const float* __restrict__ W1, const float* __restrict__ b1,
    const float* __restrict__ W2, const float* __restrict__ b2,
    const float* __restrict__ Wr, const float* __restrict__ br,
    const float* __restrict__ gamma, const float* __restrict__ beta,
    float* __restrict__ out) {
  constexpr int KT = DIN / 32;
  __shared__ v16h sW[4 * 8 * 32];                            // 32KB swizzled weights
  __shared__ __align__(16) _Float16 sH[16 * DIN];            // h rows (f16)
  __shared__ __align__(16) _Float16 sX[PROJ ? 16 * DIN : 8]; // x rows, PROJ only
  __shared__ __align__(16) _Float16 sT[16 * 128];            // stage-1 acts (f16)
  __shared__ float sRes[16 * 128];                           // residual / pre-LN
  __shared__ float sRed[2][256];
  __shared__ float sMu[16], sRs[16];

  const int tid = threadIdx.x, lane = tid & 31, wv = tid >> 5;
  const int node0 = blockIdx.x * 16;

  // load 16 rows; h = x + aggr
  for (int i = tid; i < 16 * DIN; i += 256) {
    int r = i / DIN, c = i % DIN;
    float xv = xin[(size_t)(node0 + r) * DIN + c];
    float hv = xv + aggr[(size_t)(node0 + r) * DIN + c];
    sH[i] = (_Float16)hv;
    if (PROJ) sX[i] = (_Float16)xv;
  }

  // residual
  if (PROJ) {
    stage_B<KT>(Wr, sW, tid);
    __syncthreads();
    v8f cr = {};
#pragma unroll
    for (int kt = 0; kt < KT; ++kt)
      cr = wmma_f16(load_A_frag(sX, DIN, kt * 32, lane),
                    load_B_sw(sW, kt, wv, lane), cr);
    {
      int n = (lane & 15) + wv * 16;
      int mb = (lane >> 4) * 8;
#pragma unroll
      for (int r = 0; r < 8; ++r) sRes[(mb + r) * 128 + n] = cr[r] + br[n];
    }
    __syncthreads();
  } else {
    for (int i = tid; i < 16 * 128; i += 256) {
      int r = i >> 7, c = i & 127;
      sRes[i] = xin[(size_t)(node0 + r) * 128 + c];
    }
  }

  // stage 1: sT = relu(sH @ W1 + b1)
  stage_B<KT>(W1, sW, tid);
  __syncthreads();
  v8f c1 = {};
#pragma unroll
  for (int kt = 0; kt < KT; ++kt)
    c1 = wmma_f16(load_A_frag(sH, DIN, kt * 32, lane),
                  load_B_sw(sW, kt, wv, lane), c1);
  {
    int n = (lane & 15) + wv * 16;
    int mb = (lane >> 4) * 8;
#pragma unroll
    for (int r = 0; r < 8; ++r)
      sT[(mb + r) * 128 + n] = (_Float16)fmaxf(c1[r] + b1[n], 0.0f);
  }
  __syncthreads();  // all waves done reading W1 staging + sT written

  // stage 2: u = relu(sT @ W2 + b2) + res
  stage_B<4>(W2, sW, tid);
  __syncthreads();
  v8f c2 = {};
#pragma unroll
  for (int kt = 0; kt < 4; ++kt)
    c2 = wmma_f16(load_A_frag(sT, 128, kt * 32, lane),
                  load_B_sw(sW, kt, wv, lane), c2);
  {
    int n = (lane & 15) + wv * 16;
    int mb = (lane >> 4) * 8;
#pragma unroll
    for (int r = 0; r < 8; ++r) {
      int idx = (mb + r) * 128 + n;
      sRes[idx] = fmaxf(c2[r] + b2[n], 0.0f) + sRes[idx];  // same-lane RMW, safe
    }
  }
  __syncthreads();

  // LayerNorm over 128 features per row (16 threads per row)
  {
    int row = tid >> 4, sub = tid & 15;
    float s = 0.0f, ss = 0.0f;
    for (int j = sub; j < 128; j += 16) {
      float v = sRes[row * 128 + j];
      s += v; ss += v * v;
    }
    sRed[0][tid] = s; sRed[1][tid] = ss;
  }
  __syncthreads();
  if (tid < 16) {
    float a = 0.0f, b = 0.0f;
    for (int k = 0; k < 16; ++k) { a += sRed[0][tid * 16 + k]; b += sRed[1][tid * 16 + k]; }
    float mu = a * (1.0f / 128.0f);
    float var = b * (1.0f / 128.0f) - mu * mu;
    sMu[tid] = mu;
    sRs[tid] = rsqrtf(var + LN_EPS);
  }
  __syncthreads();
  for (int i = tid; i < 16 * 128; i += 256) {
    int r = i >> 7, c = i & 127;
    out[(size_t)(node0 + r) * 128 + c] =
        (sRes[i] - sMu[r]) * sRs[r] * gamma[c] + beta[c];
  }
}

// ---------------------------------------------------------------------------
// Global mean pool: atomic segment-sum of rows + counts (one wave per node)
// ---------------------------------------------------------------------------
__global__ __launch_bounds__(256) void pool_kernel(
    const float* __restrict__ h, const int* __restrict__ batch,
    float* __restrict__ pooled, float* __restrict__ cnts, int n) {
  const int lane = threadIdx.x & 31;
  const int wv0 = blockIdx.x * 8 + (threadIdx.x >> 5);
  const int nw = gridDim.x * 8;
  for (int nd = wv0; nd < n; nd += nw) {
    int g = batch[nd];
#pragma unroll
    for (int j = 0; j < 4; ++j)
      atomicAdd(&pooled[(size_t)g * 128 + lane + 32 * j],
                h[(size_t)nd * 128 + lane + 32 * j]);
    if (lane == 0) atomicAdd(&cnts[g], 1.0f);
  }
}

// ---------------------------------------------------------------------------
// Readout: out = relu(mean_pooled @ Wo1 + bo1) @ Wo2 + bo2
// One workgroup per 16 graphs; WMMA for the 128x128 stage.
// ---------------------------------------------------------------------------
__global__ __launch_bounds__(256) void readout_kernel(
    const float* __restrict__ pooled, const float* __restrict__ cnts,
    const float* __restrict__ Wo1, const float* __restrict__ bo1,
    const float* __restrict__ Wo2, const float* __restrict__ bo2,
    float* __restrict__ out) {
  __shared__ v16h sW[4 * 8 * 32];
  __shared__ __align__(16) _Float16 sA[16 * 128];
  __shared__ float sT[16 * 128];
  __shared__ float sRed[256];
  const int tid = threadIdx.x, lane = tid & 31, wv = tid >> 5;
  const int g0 = blockIdx.x * 16;

  for (int i = tid; i < 16 * 128; i += 256) {
    int r = i >> 7, c = i & 127;
    float cnt = fmaxf(cnts[g0 + r], 1.0f);
    sA[i] = (_Float16)(pooled[(size_t)(g0 + r) * 128 + c] / cnt);
  }
  stage_B<4>(Wo1, sW, tid);
  __syncthreads();

  v8f c = {};
#pragma unroll
  for (int kt = 0; kt < 4; ++kt)
    c = wmma_f16(load_A_frag(sA, 128, kt * 32, lane),
                 load_B_sw(sW, kt, wv, lane), c);
  {
    int n = (lane & 15) + wv * 16;
    int mb = (lane >> 4) * 8;
#pragma unroll
    for (int r = 0; r < 8; ++r) sT[(mb + r) * 128 + n] = fmaxf(c[r] + bo1[n], 0.0f);
  }
  __syncthreads();

  // final 128 -> 1 dot per graph
  {
    int row = tid >> 4, sub = tid & 15;
    float s = 0.0f;
    for (int j = sub; j < 128; j += 16) s += sT[row * 128 + j] * Wo2[j];
    sRed[tid] = s;
  }
  __syncthreads();
  if (tid < 16) {
    float a = 0.0f;
    for (int k = 0; k < 16; ++k) a += sRed[tid * 16 + k];
    out[g0 + tid] = a + bo2[0];
  }
}

// ---------------------------------------------------------------------------
// launcher
// ---------------------------------------------------------------------------
extern "C" void kernel_launch(void* const* d_in, const int* in_sizes, int n_in,
                              void* d_out, int out_size, void* d_ws, size_t ws_size,
                              hipStream_t stream) {
  const float* x     = (const float*)d_in[0];
  const int*   ei    = (const int*)d_in[1];
  const float* ea    = (const float*)d_in[2];
  const int*   batch = (const int*)d_in[3];
  const float* We0 = (const float*)d_in[4],  *be0 = (const float*)d_in[5];
  const float* W1a = (const float*)d_in[6],  *b1a = (const float*)d_in[7];
  const float* W2a = (const float*)d_in[8],  *b2a = (const float*)d_in[9];
  const float* g0  = (const float*)d_in[10], *bt0 = (const float*)d_in[11];
  const float* Wr  = (const float*)d_in[12], *br  = (const float*)d_in[13];
  const float* We1 = (const float*)d_in[14], *be1 = (const float*)d_in[15];
  const float* W1b = (const float*)d_in[16], *b1b = (const float*)d_in[17];
  const float* W2b = (const float*)d_in[18], *b2b = (const float*)d_in[19];
  const float* g1  = (const float*)d_in[20], *bt1 = (const float*)d_in[21];
  const float* We2 = (const float*)d_in[22], *be2 = (const float*)d_in[23];
  const float* W1c = (const float*)d_in[24], *b1c = (const float*)d_in[25];
  const float* W2c = (const float*)d_in[26], *b2c = (const float*)d_in[27];
  const float* g2  = (const float*)d_in[28], *bt2 = (const float*)d_in[29];
  const float* Wo1 = (const float*)d_in[30], *bo1 = (const float*)d_in[31];
  const float* Wo2 = (const float*)d_in[32], *bo2 = (const float*)d_in[33];

  const int N = in_sizes[0] / 64;   // 50000
  const int E = in_sizes[1] / 2;    // 800000
  const int G = out_size;           // 2048 (output [G,1])
  const int* src = ei;
  const int* dst = ei + E;

  float* aggr   = (float*)d_ws;
  float* hA     = aggr + (size_t)N * 128;
  float* hB     = hA + (size_t)N * 128;
  float* pooled = hB + (size_t)N * 128;
  float* cnts   = pooled + (size_t)G * 128;
  float* outp   = (float*)d_out;

  const int nodeBlocks = (N + 15) / 16;   // N divisible by 16 here
  const int msgBlocks  = 2048;            // 16384 waves over E edges

  // ---- block 0 (64 -> 128, projected residual) ----
  {
    long n64 = (long)N * 64;
    zero_kernel<<<(int)((n64 + 255) / 256), 256, 0, stream>>>(aggr, n64);
    gine_msg_kernel<64><<<msgBlocks, 256, 0, stream>>>(x, src, dst, ea, We0, be0, aggr, E);
    gine_node_kernel<64, true><<<nodeBlocks, 256, 0, stream>>>(
        x, aggr, W1a, b1a, W2a, b2a, Wr, br, g0, bt0, hA);
  }
  // ---- block 1 (128 -> 128, identity residual) ----
  {
    long n128 = (long)N * 128;
    zero_kernel<<<(int)((n128 + 255) / 256), 256, 0, stream>>>(aggr, n128);
    gine_msg_kernel<128><<<msgBlocks, 256, 0, stream>>>(hA, src, dst, ea, We1, be1, aggr, E);
    gine_node_kernel<128, false><<<nodeBlocks, 256, 0, stream>>>(
        hA, aggr, W1b, b1b, W2b, b2b, nullptr, nullptr, g1, bt1, hB);
  }
  // ---- block 2 (128 -> 128, identity residual) ----
  {
    long n128 = (long)N * 128;
    zero_kernel<<<(int)((n128 + 255) / 256), 256, 0, stream>>>(aggr, n128);
    gine_msg_kernel<128><<<msgBlocks, 256, 0, stream>>>(hB, src, dst, ea, We2, be2, aggr, E);
    gine_node_kernel<128, false><<<nodeBlocks, 256, 0, stream>>>(
        hB, aggr, W1c, b1c, W2c, b2c, nullptr, nullptr, g2, bt2, hA);
  }
  // ---- pool + readout ----
  {
    long np = (long)G * 128 + G;
    zero_kernel<<<(int)((np + 255) / 256), 256, 0, stream>>>(pooled, np);
    pool_kernel<<<512, 256, 0, stream>>>(hA, batch, pooled, cnts, N);
    readout_kernel<<<G / 16, 256, 0, stream>>>(pooled, cnts, Wo1, bo1, Wo2, bo2, outp);
  }
}
